// PointCloudExtractor_51866025066719
// MI455X (gfx1250) — matrix operand order repaired and checked by reading, not compiled, not run on hardware
//
#include <hip/hip_runtime.h>

// ---------------------------------------------------------------------------
// PointCloudExtractor for MI455X (gfx1250): bf16 WMMA fused MLP pipeline
// ---------------------------------------------------------------------------

typedef __bf16 bf16;
typedef __attribute__((ext_vector_type(16))) __bf16 v16bf;
typedef __attribute__((ext_vector_type(8)))  float  v8f;

#define BN_RS 0.9995003746879121f   // 1/sqrt(1+1e-3)

// Fast SiLU: v_exp_f32 + v_add + v_rcp_f32 + v_mul (no IEEE div expansion)
__device__ __forceinline__ float silu_f(float x) {
    return x * __builtin_amdgcn_rcpf(1.0f + __expf(-x));
}

// float atomic max via sign-split integer atomics (out initialized to -inf)
__device__ __forceinline__ void atomicMaxF(float* addr, float val) {
    if (val >= 0.0f) atomicMax((int*)addr, __float_as_int(val));
    else             atomicMin((unsigned int*)addr, (unsigned int)__float_as_int(val));
}

__device__ __forceinline__ v8f wmma_bf16(v16bf a, v16bf b, v8f c) {
    // D = A(16x32 bf16) * B(32x16 bf16) + C(16x16 f32)
    return __builtin_amdgcn_wmma_f32_16x16x32_bf16(false, a, false, b, (short)0, c,
                                                   false, false);
}

// A fragment (16x32 bf16) from row-major LDS buffer, per ISA 16-bit A layout:
// lane half h: VGPR i holds k = (i>>2)*16 + h*8 + (i&3)*2 (+1)
__device__ __forceinline__ v16bf load_a_frag(const bf16* src, int sld, int rt,
                                             int kk, int lane) {
    const int nl = lane & 15, h = lane >> 4;
    const bf16* p = src + (rt * 16 + nl) * sld + (h << 3) + (kk << 5);
    v16bf a;
#pragma unroll
    for (int i = 0; i < 8; ++i) {
        int k = ((i >> 2) << 4) + ((i & 3) << 1);
        a[2 * i]     = p[k];
        a[2 * i + 1] = p[k + 1];
    }
    return a;
}

// ---------------------------------------------------------------------------
// 1) T-Net stage A: h = silu(bn(pts @ t_cw)), max over N -> xmax[B,128]
// ---------------------------------------------------------------------------
__global__ __launch_bounds__(128) void tnet1_kernel(
    const float* __restrict__ pts, const float* __restrict__ cw,
    const float* __restrict__ cb, const float* __restrict__ g1,
    const float* __restrict__ b1, float* __restrict__ xmax) {
    const int b = blockIdx.x, c = threadIdx.x;
    const float w0 = cw[c], w1 = cw[128 + c], w2 = cw[256 + c];
    const float sc = BN_RS * g1[c];
    const float sh = b1[c] + cb[c] * sc;
    const float* pb = pts + (size_t)b * 2048 * 3;
    float mx = -INFINITY;
    for (int n = 0; n < 2048; ++n) {
        float y = pb[n * 3] * w0 + pb[n * 3 + 1] * w1 + pb[n * 3 + 2] * w2;
        y = y * sc + sh;
        mx = fmaxf(mx, silu_f(y));
    }
    xmax[b * 128 + c] = mx;
}

// ---------------------------------------------------------------------------
// 2) T-Net dense: x->128->9, T matrix per batch
// ---------------------------------------------------------------------------
__global__ __launch_bounds__(128) void tnet2_kernel(
    const float* __restrict__ xmax, const float* __restrict__ d1w,
    const float* __restrict__ d1b, const float* __restrict__ g2,
    const float* __restrict__ b2, const float* __restrict__ d2w,
    const float* __restrict__ d2b, float* __restrict__ Tm) {
    __shared__ float xs[128], x2[128];
    const int b = blockIdx.x, c = threadIdx.x;
    xs[c] = xmax[b * 128 + c];
    __syncthreads();
    float acc = 0.f;
    for (int i = 0; i < 128; ++i) acc += xs[i] * d1w[i * 128 + c];
    const float sc = BN_RS * g2[c];
    float y = acc * sc + (d1b[c] * sc + b2[c]);
    x2[c] = silu_f(y);
    __syncthreads();
    if (c < 9) {
        float a = 0.f;
        for (int i = 0; i < 128; ++i) a += x2[i] * d2w[i * 9 + c];
        Tm[b * 9 + c] = a + d2b[c];
    }
}

// ---------------------------------------------------------------------------
// 3) pc = pts @ T
// ---------------------------------------------------------------------------
__global__ __launch_bounds__(256) void transform_kernel(
    const float* __restrict__ pts, const float* __restrict__ Tm,
    float* __restrict__ pc) {
    const int idx = blockIdx.x * 256 + threadIdx.x;
    if (idx >= 4 * 2048) return;
    const int b = idx >> 11;
    const float* Tb = Tm + b * 9;
    const float* pp = pts + (size_t)idx * 3;
    const float x = pp[0], y = pp[1], z = pp[2];
    float* o = pc + (size_t)idx * 3;
    o[0] = x * Tb[0] + y * Tb[3] + z * Tb[6];
    o[1] = x * Tb[1] + y * Tb[4] + z * Tb[7];
    o[2] = x * Tb[2] + y * Tb[5] + z * Tb[8];
}

// ---------------------------------------------------------------------------
// 4) Multi-scale grouping: one wave per point (K=32 == wave32).
//    top_k on a 0/1 mask == first in-radius indices ascending, then first
//    out-of-radius indices ascending.  f9 rows stored as bf16[16] (9 + pad).
// ---------------------------------------------------------------------------
__global__ __launch_bounds__(256) void group_kernel(
    const float* __restrict__ pc, unsigned short* __restrict__ f9u) {
    __shared__ unsigned short oneIdx[8][3][32];
    __shared__ unsigned short zeroIdx[8][3][32];
    bf16* f9 = (bf16*)f9u;
    const int lane = threadIdx.x & 31;
    const int w    = threadIdx.x >> 5;
    const int p    = blockIdx.x * 8 + w;
    const int b = p >> 11, n = p & 2047;
    const float* pcb = pc + (size_t)b * 2048 * 3;
    const float cx = pcb[n * 3], cy = pcb[n * 3 + 1], cz = pcb[n * 3 + 2];
    const float r2[3] = {0.01f, 0.09f, 0.25f};
    int cnt1[3] = {0, 0, 0}, cntz[3] = {0, 0, 0};
    for (int c = 0; c < 64; ++c) {
        const int m = c * 32 + lane;
        const float dx = pcb[m * 3] - cx;
        const float dy = pcb[m * 3 + 1] - cy;
        const float dz = pcb[m * 3 + 2] - cz;
        const float d2 = dx * dx + dy * dy + dz * dz;
        const unsigned below = (1u << lane) - 1u;
#pragma unroll
        for (int r = 0; r < 3; ++r) {
            const bool in = d2 <= r2[r];
            const unsigned mask = __builtin_amdgcn_ballot_w32(in);
            if (in && cnt1[r] < 32) {
                int pos = cnt1[r] + __popc(mask & below);
                if (pos < 32) oneIdx[w][r][pos] = (unsigned short)m;
            }
            if (!in && cntz[r] < 32) {
                int pos = cntz[r] + __popc((~mask) & below);
                if (pos < 32) zeroIdx[w][r][pos] = (unsigned short)m;
            }
            cnt1[r] += __popc(mask);
            cntz[r] += __popc(~mask);
        }
    }
    __syncthreads();
    const int k = lane;
    const size_t rowOff = ((size_t)p * 32 + k) * 16;
#pragma unroll
    for (int r = 0; r < 3; ++r) {
        const int c1  = cnt1[r];
        const int sel = (k < c1) ? oneIdx[w][r][k] : zeroIdx[w][r][k - c1];
        f9[rowOff + 3 * r + 0] = (bf16)pcb[sel * 3 + 0];
        f9[rowOff + 3 * r + 1] = (bf16)pcb[sel * 3 + 1];
        f9[rowOff + 3 * r + 2] = (bf16)pcb[sel * 3 + 2];
    }
#pragma unroll
    for (int j = 9; j < 16; ++j) f9[rowOff + j] = (bf16)0.0f;
}

// ---------------------------------------------------------------------------
// 5) Weight packing: fp32 [K][N] row-major -> bf16 WMMA B-fragment-linear
//    layout (lane = N column, lane-half = K half, 16 contiguous bf16 / lane)
// ---------------------------------------------------------------------------
__global__ __launch_bounds__(256) void pack_w_kernel(
    const float* __restrict__ src, unsigned short* __restrict__ dstu,
    int K, int N, int Kp, int Np) {
    bf16* dst = (bf16*)dstu;
    const int p = blockIdx.x * 256 + threadIdx.x;
    if (p >= Kp * Np) return;
    const int e    = p & 15;
    const int lane = (p >> 4) & 31;
    const int rest = p >> 9;
    const int nCT  = Np >> 4;
    const int ct   = rest % nCT;
    const int kk   = rest / nCT;
    const int k = (kk << 5) + ((e >> 3) << 4) + ((lane >> 4) << 3) + (e & 7);
    const int n = (ct << 4) + (lane & 15);
    const float v = (k < K && n < N) ? src[k * N + n] : 0.0f;
    dst[p] = (bf16)v;
}

// ---------------------------------------------------------------------------
// 6) Output init to -inf (max identity)
// ---------------------------------------------------------------------------
__global__ __launch_bounds__(256) void init_out_kernel(float* __restrict__ out, int n) {
    const int i = blockIdx.x * 256 + threadIdx.x;
    if (i < n) out[i] = -INFINITY;
}

// ---------------------------------------------------------------------------
// 7) Fused MLP over 64-row tiles, bf16 WMMA, LDS-resident activations
// ---------------------------------------------------------------------------
template <int MODE>  // 0 = bias+BN+SiLU -> LDS,  1 = residual add -> LDS
__device__ void gemm_layer(const bf16* src, int sld, bf16* dst, int dld,
                           const bf16* Wp, int Kp, int Np,
                           const float* bias, const float* g, const float* be) {
    const int lane = threadIdx.x & 31;
    const int wave = threadIdx.x >> 5;
    const int nCT = Np >> 4;
    const int nK  = Kp >> 5;
    const int nT  = 4 * nCT;
    const int nl = lane & 15, h = lane >> 4;
    for (int t = wave; t < nT; t += 8) {
        const int rt = t & 3, ct = t >> 2;
        v8f acc = {0.f, 0.f, 0.f, 0.f, 0.f, 0.f, 0.f, 0.f};
        for (int kk = 0; kk < nK; ++kk) {
            v16bf a  = load_a_frag(src, sld, rt, kk, lane);
            v16bf bm = *(const v16bf*)(Wp + (((size_t)kk * nCT + ct) * 32 + lane) * 16);
            acc = wmma_bf16(a, bm, acc);
        }
        const int n = ct * 16 + nl;
        float sc = 1.f, sh = 0.f;
        if constexpr (MODE == 0) {
            sc = BN_RS * g[n];
            sh = be[n] + bias[n] * sc;
        }
#pragma unroll
        for (int r = 0; r < 8; ++r) {
            const int m = rt * 16 + h * 8 + r;
            float v = acc[r];
            if constexpr (MODE == 1) {
                v += bias[n] + (float)src[m * sld + n];
            } else {
                v = silu_f(v * sc + sh);
            }
            dst[m * dld + n] = (bf16)v;
        }
    }
}

// Final layer: 4 row-tiles per B fragment (B reuse), fold N-max via atomics.
__device__ void gemm_final(const bf16* src, int sld, const bf16* Wp, int Kp, int Np,
                           const float* bias, const float* g, const float* be,
                           int rowBase, float* out) {
    const int lane = threadIdx.x & 31;
    const int wave = threadIdx.x >> 5;
    const int nCT = Np >> 4;   // 11
    const int nK  = Kp >> 5;   // 8
    const int nl = lane & 15, h = lane >> 4;
    const int b = rowBase >> 16;            // 64 rows never straddle a batch
    for (int ct = wave; ct < nCT; ct += 8) {
        v8f a0 = {0.f,0.f,0.f,0.f,0.f,0.f,0.f,0.f};
        v8f a1 = a0, a2 = a0, a3 = a0;
        for (int kk = 0; kk < nK; ++kk) {
            v16bf bm = *(const v16bf*)(Wp + (((size_t)kk * nCT + ct) * 32 + lane) * 16);
            a0 = wmma_bf16(load_a_frag(src, sld, 0, kk, lane), bm, a0);
            a1 = wmma_bf16(load_a_frag(src, sld, 1, kk, lane), bm, a1);
            a2 = wmma_bf16(load_a_frag(src, sld, 2, kk, lane), bm, a2);
            a3 = wmma_bf16(load_a_frag(src, sld, 3, kk, lane), bm, a3);
        }
        const int n = ct * 16 + nl;
        if (n >= 170) continue;             // pad columns
        const float sc = BN_RS * g[n];
        const float sh = be[n] + bias[n] * sc;
#pragma unroll
        for (int r = 0; r < 8; ++r) {
            const int k0 = h * 8 + r;        // k index for row-tiles 0/2
            const int k1 = 16 + h * 8 + r;   // k index for row-tiles 1/3
            const float v0 = silu_f(a0[r] * sc + sh);
            const float v1 = silu_f(a1[r] * sc + sh);
            const float v2 = silu_f(a2[r] * sc + sh);
            const float v3 = silu_f(a3[r] * sc + sh);
            atomicMaxF(out + b * 5440 + k0 * 170 + n, fmaxf(v0, v2));
            atomicMaxF(out + b * 5440 + k1 * 170 + n, fmaxf(v1, v3));
        }
    }
}

__global__ __launch_bounds__(256) void mlp_kernel(
    const unsigned short* __restrict__ f9u,
    const unsigned short* __restrict__ c1pu, const unsigned short* __restrict__ respu,
    const unsigned short* __restrict__ b0pu, const unsigned short* __restrict__ b1pu,
    const unsigned short* __restrict__ b2pu,
    const float* __restrict__ c1_b, const float* __restrict__ c1_g,
    const float* __restrict__ c1_be, const float* __restrict__ res_b,
    const float* __restrict__ b0_b, const float* __restrict__ b0_g,
    const float* __restrict__ b0_be, const float* __restrict__ b1_b,
    const float* __restrict__ b1_g, const float* __restrict__ b1_be,
    const float* __restrict__ b2_b, const float* __restrict__ b2_g,
    const float* __restrict__ b2_be, float* __restrict__ out) {
    extern __shared__ char smem_raw[];
    bf16* inA  = (bf16*)smem_raw;        // 64 x 32
    bf16* act1 = inA + 64 * 32;          // 64 x 512
    bf16* act2 = act1 + 64 * 512;        // 64 x 512
    const bf16* f9 = (const bf16*)f9u;
    const int tid = threadIdx.x;
    const int rowBase = blockIdx.x * 64;
    {   // stage padded input rows into LDS
        const int row = tid >> 2, q = tid & 3;
        const bf16* s = f9 + ((size_t)(rowBase + row) << 4);
        bf16* d = inA + row * 32;
#pragma unroll
        for (int j = 0; j < 8; ++j) {
            const int col = q * 8 + j;
            d[col] = (q < 2) ? s[col] : (bf16)0.0f;
        }
    }
    __syncthreads();
    gemm_layer<0>(inA, 32, act1, 512, (const bf16*)c1pu, 32, 512, c1_b, c1_g, c1_be);
    __syncthreads();
    gemm_layer<1>(act1, 512, act2, 512, (const bf16*)respu, 512, 512, res_b, nullptr, nullptr);
    __syncthreads();
    gemm_layer<0>(act2, 512, act1, 512, (const bf16*)b0pu, 512, 512, b0_b, b0_g, b0_be);
    __syncthreads();
    gemm_layer<0>(act1, 512, act2, 512, (const bf16*)b1pu, 512, 256, b1_b, b1_g, b1_be);
    __syncthreads();
    gemm_final(act2, 512, (const bf16*)b2pu, 256, 176, b2_b, b2_g, b2_be, rowBase, out);
}

// ---------------------------------------------------------------------------
// Host launch
// ---------------------------------------------------------------------------
extern "C" void kernel_launch(void* const* d_in, const int* in_sizes, int n_in,
                              void* d_out, int out_size, void* d_ws, size_t ws_size,
                              hipStream_t stream) {
    (void)in_sizes; (void)n_in; (void)ws_size;
    const float* pts   = (const float*)d_in[0];
    const float* t_cw  = (const float*)d_in[1];
    const float* t_cb  = (const float*)d_in[2];
    const float* t_g1  = (const float*)d_in[3];
    const float* t_b1  = (const float*)d_in[4];
    const float* t_d1w = (const float*)d_in[5];
    const float* t_d1b = (const float*)d_in[6];
    const float* t_g2  = (const float*)d_in[7];
    const float* t_b2  = (const float*)d_in[8];
    const float* t_d2w = (const float*)d_in[9];
    const float* t_d2b = (const float*)d_in[10];
    const float* c1_w  = (const float*)d_in[11];
    const float* c1_b  = (const float*)d_in[12];
    const float* c1_g  = (const float*)d_in[13];
    const float* c1_be = (const float*)d_in[14];
    const float* res_w = (const float*)d_in[15];
    const float* res_b = (const float*)d_in[16];
    const float* b0_w  = (const float*)d_in[17];
    const float* b0_b  = (const float*)d_in[18];
    const float* b0_g  = (const float*)d_in[19];
    const float* b0_be = (const float*)d_in[20];
    const float* b1_w  = (const float*)d_in[21];
    const float* b1_b  = (const float*)d_in[22];
    const float* b1_g  = (const float*)d_in[23];
    const float* b1_be = (const float*)d_in[24];
    const float* b2_w  = (const float*)d_in[25];
    const float* b2_b  = (const float*)d_in[26];
    const float* b2_g  = (const float*)d_in[27];
    const float* b2_be = (const float*)d_in[28];
    float* out = (float*)d_out;

    char* ws = (char*)d_ws;
    size_t off = 0;
    auto alloc = [&](size_t bytes) -> char* {
        char* p = ws + off;
        off += (bytes + 255) & ~(size_t)255;
        return p;
    };
    float*          xmax = (float*)alloc((size_t)4 * 128 * 4);
    float*          Tm   = (float*)alloc((size_t)4 * 9 * 4);
    float*          pc   = (float*)alloc((size_t)4 * 2048 * 3 * 4);
    unsigned short* f9   = (unsigned short*)alloc((size_t)262144 * 16 * 2);
    unsigned short* c1p  = (unsigned short*)alloc((size_t)32 * 512 * 2);
    unsigned short* resp = (unsigned short*)alloc((size_t)512 * 512 * 2);
    unsigned short* b0p  = (unsigned short*)alloc((size_t)512 * 512 * 2);
    unsigned short* b1p  = (unsigned short*)alloc((size_t)512 * 256 * 2);
    unsigned short* b2p  = (unsigned short*)alloc((size_t)256 * 176 * 2);

    init_out_kernel<<<(out_size + 255) / 256, 256, 0, stream>>>(out, out_size);
    tnet1_kernel<<<4, 128, 0, stream>>>(pts, t_cw, t_cb, t_g1, t_b1, xmax);
    tnet2_kernel<<<4, 128, 0, stream>>>(xmax, t_d1w, t_d1b, t_g2, t_b2, t_d2w, t_d2b, Tm);
    transform_kernel<<<32, 256, 0, stream>>>(pts, Tm, pc);
    group_kernel<<<1024, 256, 0, stream>>>(pc, f9);
    pack_w_kernel<<<(32 * 512 + 255) / 256, 256, 0, stream>>>(c1_w, c1p, 9, 512, 32, 512);
    pack_w_kernel<<<(512 * 512) / 256, 256, 0, stream>>>(res_w, resp, 512, 512, 512, 512);
    pack_w_kernel<<<(512 * 512) / 256, 256, 0, stream>>>(b0_w, b0p, 512, 512, 512, 512);
    pack_w_kernel<<<(512 * 256) / 256, 256, 0, stream>>>(b1_w, b1p, 512, 256, 512, 256);
    pack_w_kernel<<<(256 * 176) / 256, 256, 0, stream>>>(b2_w, b2p, 256, 170, 256, 176);

    const size_t lds_bytes = (size_t)(64 * 32 + 2 * 64 * 512) * sizeof(unsigned short);
    mlp_kernel<<<4096, 256, lds_bytes, stream>>>(
        f9, c1p, resp, b0p, b1p, b2p,
        c1_b, c1_g, c1_be, res_b,
        b0_b, b0_g, b0_be, b1_b, b1_g, b1_be,
        b2_b, b2_g, b2_be, out);
}
